// ParallelTransformerBlock_16114717294991
// MI455X (gfx1250) — matrix operand order repaired and checked by reading, not compiled, not run on hardware
//
#include <hip/hip_runtime.h>
#include <hip/hip_bf16.h>

// ---------------- problem constants ----------------
#define DIMX 2048
#define HEADS 16
#define DHEAD 128
#define NSEQ 2048
#define BATCH 2
#define ATTN_INNER 2048          // HEADS*DHEAD
#define FF_INNER 8192            // DIM*4
#define FUSEDN 18688             // ATTN_INNER + 2*DHEAD + 2*FF_INNER
#define MROWS 4096               // BATCH*NSEQ

typedef __attribute__((ext_vector_type(16))) __bf16 v16bf;
typedef __attribute__((ext_vector_type(8)))  __bf16 v8bf;
typedef __attribute__((ext_vector_type(8)))  float  v8f;
typedef __attribute__((ext_vector_type(4)))  int    v4i;

#define AS1 __attribute__((address_space(1)))
#define AS3 __attribute__((address_space(3)))

#if __has_builtin(__builtin_amdgcn_global_load_async_to_lds_b128)
#define HAVE_ASYNC_LDS 1
#endif

// ---------------- async copy helpers ----------------
__device__ __forceinline__ void async_copy_b128(const __bf16* g, __bf16* l) {
#if defined(HAVE_ASYNC_LDS)
  __builtin_amdgcn_global_load_async_to_lds_b128(
      (AS1 v4i*)(AS1 const void*)g,
      (AS3 v4i*)(AS3 void*)l,
      0, 0);
#else
  *(v8bf*)l = *(const v8bf*)g;     // synchronous fallback
#endif
}

__device__ __forceinline__ void wait_async_0() {
#if defined(HAVE_ASYNC_LDS)
# if __has_builtin(__builtin_amdgcn_s_wait_asynccnt)
  __builtin_amdgcn_s_wait_asynccnt(0);
# else
  asm volatile("s_wait_asynccnt 0x0" ::: "memory");
# endif
#endif
}

__device__ __forceinline__ void wait_async_2() {
#if defined(HAVE_ASYNC_LDS)
# if __has_builtin(__builtin_amdgcn_s_wait_asynccnt)
  __builtin_amdgcn_s_wait_asynccnt(2);
# else
  asm volatile("s_wait_asynccnt 0x2" ::: "memory");
# endif
#endif
}

// ---------------- WMMA fragment helpers ----------------
// A fragment: 16x32 bf16, lane = M row (lane&15); lanes<16 hold K {0..7,16..23},
// lanes>=16 hold K {8..15,24..31} (ISA 7.12.2 16-bit A table).
__device__ __forceinline__ v16bf load_a_frag(const __bf16* __restrict__ A, int lda,
                                             int kBase, int lane) {
  const int r   = lane & 15;
  const int klo = (lane >> 4) << 3;            // 0 or 8
  const __bf16* p = A + (size_t)r * lda + kBase + klo;
  v8bf lo = *(const v8bf*)p;                   // K klo..klo+7
  v8bf hi = *(const v8bf*)(p + 16);            // K 16+klo..16+klo+7
  v16bf f;
#pragma unroll
  for (int i = 0; i < 8; ++i) { f[i] = lo[i]; f[8 + i] = hi[i]; }
  return f;
}

// B fragment: 32x16 bf16, lane = N column (lane&15); lanes<16 hold K {0..7,16..23},
// lanes>=16 hold K {8..15,24..31}.
__device__ __forceinline__ v16bf load_b_frag(const __bf16* __restrict__ B, int ldb,
                                             int kBase, int nBase, int lane) {
  const int klo = (lane >> 4) << 3;
  const __bf16* p = B + (size_t)(kBase + klo) * ldb + nBase + (lane & 15);
  v16bf f;
#pragma unroll
  for (int i = 0; i < 8; ++i) { f[i] = p[(size_t)i * ldb]; f[8 + i] = p[(size_t)(16 + i) * ldb]; }
  return f;
}

// B fragment straight out of an LDS tile row-major [32][128]
__device__ __forceinline__ v16bf load_b_frag_lds(const __bf16* tile, int nOff, int lane) {
  const int klo = (lane >> 4) << 3;
  const __bf16* p = tile + klo * 128 + nOff + (lane & 15);
  v16bf f;
#pragma unroll
  for (int i = 0; i < 8; ++i) { f[i] = p[i * 128]; f[8 + i] = p[(16 + i) * 128]; }
  return f;
}

__device__ __forceinline__ v8f wmma_bf16(v16bf a, v16bf b, v8f c) {
  return __builtin_amdgcn_wmma_f32_16x16x32_bf16(false, a, false, b, (short)0, c,
                                                 false, false);
}

// ---------------- small kernels ----------------
__global__ void convert_bf16_kernel(const float* __restrict__ src,
                                    __bf16* __restrict__ dst, size_t n) {
  size_t i = (size_t)blockIdx.x * blockDim.x + threadIdx.x;
  size_t stride = (size_t)gridDim.x * blockDim.x;
  for (; i < n; i += stride) dst[i] = (__bf16)src[i];
}

__global__ __launch_bounds__(256) void layernorm_kernel(const float* __restrict__ x,
                                                        const float* __restrict__ gamma,
                                                        __bf16* __restrict__ xn) {
  __shared__ float ssum[256], ssq[256];
  const int row = blockIdx.x, tid = threadIdx.x;
  const float* xr = x + (size_t)row * DIMX;
  float s = 0.f, q = 0.f, vals[8];
#pragma unroll
  for (int j = 0; j < 8; ++j) {
    float v = xr[tid + j * 256];
    vals[j] = v; s += v; q += v * v;
  }
  ssum[tid] = s; ssq[tid] = q;
  __syncthreads();
  for (int off = 128; off > 0; off >>= 1) {
    if (tid < off) { ssum[tid] += ssum[tid + off]; ssq[tid] += ssq[tid + off]; }
    __syncthreads();
  }
  const float mean = ssum[0] * (1.0f / DIMX);
  const float var  = ssq[0] * (1.0f / DIMX) - mean * mean;
  const float rstd = rsqrtf(var + 1e-5f);
#pragma unroll
  for (int j = 0; j < 8; ++j) {
    int c = tid + j * 256;
    xn[(size_t)row * DIMX + c] = (__bf16)((vals[j] - mean) * rstd * gamma[c]);
  }
}

// RoPE on q (plus 1/sqrt(DHEAD) scale) and k; q rotated (bf16, row-major),
// k rotated + transposed to [B, DHEAD, NSEQ] (bf16), v converted to bf16.
__global__ __launch_bounds__(256) void rope_kernel(const float* __restrict__ qf,
                                                   const float* __restrict__ kf,
                                                   const float* __restrict__ vf,
                                                   __bf16* __restrict__ qr,
                                                   __bf16* __restrict__ kT,
                                                   __bf16* __restrict__ vb) {
  const int row = blockIdx.x;          // 0..MROWS-1
  const int b = row / NSEQ, n = row % NSEQ;
  const int tid = threadIdx.x;
  const float LN1E4_OVER_64 = 9.210340371976184f / 64.0f;  // ln(10000)/64
  const float qscale = 0.08838834764831845f;               // 128^-0.5
#pragma unroll
  for (int j = 0; j < 8; ++j) {
    int c = tid + j * 256;
    int base = c & ~127;
    int dd = c & 127;
    float t = qf[(size_t)row * ATTN_INNER + c];
    float val;
    if (dd < 64) {
      float f = (float)n * __expf(-(float)dd * LN1E4_OVER_64);
      float partner = qf[(size_t)row * ATTN_INNER + base + dd + 64];
      val = t * __cosf(f) - partner * __sinf(f);
    } else {
      int d0 = dd - 64;
      float f = (float)n * __expf(-(float)d0 * LN1E4_OVER_64);
      float partner = qf[(size_t)row * ATTN_INNER + base + d0];
      val = t * __cosf(f) + partner * __sinf(f);
    }
    qr[(size_t)row * ATTN_INNER + c] = (__bf16)(val * qscale);
  }
  if (tid < DHEAD) {
    int dd = tid;
    float t = kf[(size_t)row * DHEAD + dd];
    float val;
    if (dd < 64) {
      float f = (float)n * __expf(-(float)dd * LN1E4_OVER_64);
      val = t * __cosf(f) - kf[(size_t)row * DHEAD + dd + 64] * __sinf(f);
    } else {
      int d0 = dd - 64;
      float f = (float)n * __expf(-(float)d0 * LN1E4_OVER_64);
      val = t * __cosf(f) + kf[(size_t)row * DHEAD + d0] * __sinf(f);
    }
    kT[((size_t)b * DHEAD + dd) * NSEQ + n] = (__bf16)val;
  } else {
    int d = tid - DHEAD;
    vb[(size_t)row * DHEAD + d] = (__bf16)vf[(size_t)row * DHEAD + d];
  }
}

__global__ void swiglu_kernel(__bf16* __restrict__ ff) {
  size_t idx = (size_t)blockIdx.x * blockDim.x + threadIdx.x;
  const size_t total = (size_t)MROWS * FF_INNER;
  const size_t stride = (size_t)gridDim.x * blockDim.x;
  for (; idx < total; idx += stride) {
    size_t row = idx / FF_INNER;
    int j = (int)(idx % FF_INNER);
    size_t base = row * (size_t)(2 * FF_INNER);
    float xv = (float)ff[base + j];
    float g  = (float)ff[base + FF_INNER + j];
    float sg = g / (1.0f + __expf(-g));      // silu(gate)
    ff[base + j] = (__bf16)(sg * xv);        // in-place over ff_x half
  }
}

// =============================================================================
// GEMM with async-to-LDS double-buffered B staging.
// block = 256 threads (8 wave32); block tile 128(M) x 128(N).
// Wave w owns rows [w*16, w*16+16), all 128 N columns (8 f32 accumulators).
// B k-tile (32x128 bf16 = 8 KB) is staged in LDS once per block per k-step via
// GLOBAL_LOAD_ASYNC_TO_LDS_B128 (ASYNCcnt), double buffered.
// All 8 B fragments are pulled into registers first so the 8 WMMAs issue
// back-to-back with a single dscnt wait instead of stalling per-WMMA.
// =============================================================================
__device__ __forceinline__ void gemm_core_run(const __bf16* __restrict__ A, int lda,
                                              const __bf16* __restrict__ B, int ldb,
                                              int K, int nBase, int mBase,
                                              __bf16 (*Bt)[32][128], v8f acc[8]) {
  const int tid = threadIdx.x;
  const int lane = tid & 31;
  const __bf16* A0 = A + (size_t)mBase * lda;

  // async staging map: thread t copies 16B chunks at (row=t/16, col=(t%16)*8)
  // and (row=t/16+16, same col) of the 32x128 tile.
  const int crow = tid >> 4;         // 0..15
  const int ccol = (tid & 15) * 8;   // 0..120

  const int steps = K >> 5;
  {  // issue first tile
    const __bf16* g0 = B + (size_t)crow * ldb + nBase + ccol;
    async_copy_b128(g0, &Bt[0][crow][ccol]);
    async_copy_b128(g0 + (size_t)16 * ldb, &Bt[0][crow + 16][ccol]);
  }
  for (int s = 0; s < steps; ++s) {
    const int cur = s & 1;
    if (s + 1 < steps) {
      const int k = (s + 1) << 5;
      const __bf16* g0 = B + (size_t)(k + crow) * ldb + nBase + ccol;
      async_copy_b128(g0, &Bt[cur ^ 1][crow][ccol]);
      async_copy_b128(g0 + (size_t)16 * ldb, &Bt[cur ^ 1][crow + 16][ccol]);
      wait_async_2();                 // first 2 (cur tile) retired, in order
    } else {
      wait_async_0();
    }
    __syncthreads();                  // everyone's slice of cur tile visible
    __builtin_prefetch(A0 + (size_t)(lane & 15) * lda + (s << 5) + 64, 0, 1);
    v16bf a = load_a_frag(A0, lda, s << 5, lane);
    // batch all 8 B fragments, then burst the WMMAs
    v16bf bfrag[8];
#pragma unroll
    for (int t = 0; t < 8; ++t)
      bfrag[t] = load_b_frag_lds(&Bt[cur][0][0], t * 16, lane);
#pragma unroll
    for (int t = 0; t < 8; ++t)
      acc[t] = wmma_bf16(a, bfrag[t], acc[t]);
    __syncthreads();                  // done reading cur before it is overwritten
  }
}

__global__ __launch_bounds__(256) void gemm_bf16_kernel(const __bf16* __restrict__ A, int lda,
                                                        const __bf16* __restrict__ B, int ldb,
                                                        float* __restrict__ C, int ldc,
                                                        int K, int accumulate) {
  __shared__ __align__(16) __bf16 Bt[2][32][128];   // 16 KB
  const int lane = threadIdx.x & 31;
  const int wave = threadIdx.x >> 5;
  const int nBase = blockIdx.x * 128;
  const int mBase = blockIdx.y * 128 + wave * 16;
  v8f acc[8] = {};
  gemm_core_run(A, lda, B, ldb, K, nBase, mBase, Bt, acc);
  const int n15 = lane & 15;
  const int rg = (lane >> 4) << 3;                  // C layout row group
#pragma unroll
  for (int t = 0; t < 8; ++t)
#pragma unroll
    for (int i = 0; i < 8; ++i) {
      size_t idx = (size_t)(mBase + rg + i) * ldc + nBase + t * 16 + n15;
      C[idx] = accumulate ? (C[idx] + acc[t][i]) : acc[t][i];
    }
}

__global__ __launch_bounds__(256) void gemm_fused_kernel(const __bf16* __restrict__ A,
                                                         const __bf16* __restrict__ B,
                                                         float* __restrict__ qf,
                                                         float* __restrict__ kf,
                                                         float* __restrict__ vf,
                                                         __bf16* __restrict__ ff) {
  __shared__ __align__(16) __bf16 Bt[2][32][128];
  const int lane = threadIdx.x & 31;
  const int wave = threadIdx.x >> 5;
  const int nBase = blockIdx.x * 128;
  const int mBase = blockIdx.y * 128 + wave * 16;
  v8f acc[8] = {};
  gemm_core_run(A, DIMX, B, FUSEDN, DIMX, nBase, mBase, Bt, acc);
  const int n15 = lane & 15;
  const int rg = (lane >> 4) << 3;
#pragma unroll
  for (int t = 0; t < 8; ++t)
#pragma unroll
    for (int i = 0; i < 8; ++i) {
      int row = mBase + rg + i;
      int col = nBase + t * 16 + n15;
      float v = acc[t][i];
      if (col < ATTN_INNER)
        qf[(size_t)row * ATTN_INNER + col] = v;
      else if (col < ATTN_INNER + DHEAD)
        kf[(size_t)row * DHEAD + (col - ATTN_INNER)] = v;
      else if (col < ATTN_INNER + 2 * DHEAD)
        vf[(size_t)row * DHEAD + (col - ATTN_INNER - DHEAD)] = v;
      else
        ff[(size_t)row * (2 * FF_INNER) + (col - ATTN_INNER - 2 * DHEAD)] = (__bf16)v;
    }
}

// ---------------- causal flash attention (MQA), WMMA bf16 ---------------------
// grid = (NSEQ/64, HEADS, BATCH), block = 128 (4 waves; each wave owns 16 q rows)
__global__ __launch_bounds__(128) void attn_kernel(const __bf16* __restrict__ qr,
                                                   const __bf16* __restrict__ kT,
                                                   const __bf16* __restrict__ vb,
                                                   __bf16* __restrict__ attn) {
  __shared__ __bf16 Plds[4][16][64];         // wave-private P staging (8 KB)
  const int lane = threadIdx.x & 31;
  const int wave = threadIdx.x >> 5;
  const int qBlk = blockIdx.x;               // 64-row query block
  const int h = blockIdx.y;
  const int b = blockIdx.z;
  const int r0 = qBlk * 64 + wave * 16;      // sequence-local base row of this wave
  const int n15 = lane & 15;
  const int rg = (lane >> 4) << 3;
  const int klo = rg;

  v16bf qa[4];
  {
    const __bf16* Abase = qr + ((size_t)(b * NSEQ + r0)) * ATTN_INNER + h * DHEAD;
#pragma unroll
    for (int dk = 0; dk < 4; ++dk) qa[dk] = load_a_frag(Abase, ATTN_INNER, dk * 32, lane);
  }

  v8f o[8] = {};
  float mrow[8], lrow[8];
#pragma unroll
  for (int i = 0; i < 8; ++i) { mrow[i] = -3.0e38f; lrow[i] = 0.f; }

  const __bf16* kTb = kT + (size_t)b * DHEAD * NSEQ;

  for (int jb = 0; jb <= qBlk; ++jb) {
    const int jBase = jb * 64;
    v8f s[4] = {};
#pragma unroll
    for (int dk = 0; dk < 4; ++dk) {
#pragma unroll
      for (int t = 0; t < 4; ++t) {
        v16bf bk = load_b_frag(kTb, NSEQ, dk * 32, jBase + t * 16, lane);
        s[t] = wmma_bf16(qa[dk], bk, s[t]);
      }
    }
    float alpha[8];
#pragma unroll
    for (int i = 0; i < 8; ++i) {
      const int rowg = r0 + rg + i;
#pragma unroll
      for (int t = 0; t < 4; ++t) {
        int colg = jBase + t * 16 + n15;
        if (colg > rowg) s[t][i] = -1e30f;
      }
      float mx = fmaxf(fmaxf(s[0][i], s[1][i]), fmaxf(s[2][i], s[3][i]));
#pragma unroll
      for (int off = 1; off < 16; off <<= 1) mx = fmaxf(mx, __shfl_xor(mx, off, 32));
      float mn = fmaxf(mrow[i], mx);
      alpha[i] = __expf(mrow[i] - mn);
      mrow[i] = mn;
      float ls = 0.f;
#pragma unroll
      for (int t = 0; t < 4; ++t) {
        float p = __expf(s[t][i] - mn);
        s[t][i] = p;
        ls += p;
      }
#pragma unroll
      for (int off = 1; off < 16; off <<= 1) ls += __shfl_xor(ls, off, 32);
      lrow[i] = lrow[i] * alpha[i] + ls;
#pragma unroll
      for (int dt = 0; dt < 8; ++dt) o[dt][i] *= alpha[i];
    }
#pragma unroll
    for (int t = 0; t < 4; ++t)
#pragma unroll
      for (int i = 0; i < 8; ++i)
        Plds[wave][rg + i][t * 16 + n15] = (__bf16)s[t][i];
    asm volatile("s_wait_dscnt 0" ::: "memory");   // wave-synchronous LDS visibility
#pragma unroll
    for (int ks = 0; ks < 2; ++ks) {
      const __bf16* p = &Plds[wave][n15][ks * 32 + klo];
      v8bf lo = *(const v8bf*)p;
      v8bf hi = *(const v8bf*)(p + 16);
      v16bf pa;
#pragma unroll
      for (int i = 0; i < 8; ++i) { pa[i] = lo[i]; pa[8 + i] = hi[i]; }
      const __bf16* Vbase = vb + ((size_t)(b * NSEQ + jBase + ks * 32)) * DHEAD;
#pragma unroll
      for (int dt = 0; dt < 8; ++dt) {
        v16bf bv = load_b_frag(Vbase, DHEAD, 0, dt * 16, lane);
        o[dt] = wmma_bf16(pa, bv, o[dt]);
      }
    }
  }
#pragma unroll
  for (int i = 0; i < 8; ++i) {
    float inv = 1.0f / lrow[i];
    int rowg = b * NSEQ + r0 + rg + i;
#pragma unroll
    for (int dt = 0; dt < 8; ++dt)
      attn[(size_t)rowg * ATTN_INNER + h * DHEAD + dt * 16 + n15] =
          (__bf16)(o[dt][i] * inv);
  }
}

// ---------------- launch ------------------------------------------------------
extern "C" void kernel_launch(void* const* d_in, const int* in_sizes, int n_in,
                              void* d_out, int out_size, void* d_ws, size_t ws_size,
                              hipStream_t stream) {
  (void)in_sizes; (void)n_in; (void)out_size; (void)ws_size;
  const float* x          = (const float*)d_in[0];
  const float* gamma      = (const float*)d_in[1];
  const float* w_fused    = (const float*)d_in[2];
  const float* w_attn_out = (const float*)d_in[3];
  const float* w_ff_out   = (const float*)d_in[4];
  float* out = (float*)d_out;

  char* ws = (char*)d_ws;
  size_t off = 0;
  auto alloc = [&](size_t bytes) -> void* {
    void* p = ws + off;
    off += (bytes + 255) & ~(size_t)255;
    return p;
  };
  __bf16* wf   = (__bf16*)alloc((size_t)DIMX * FUSEDN * 2);
  __bf16* wao  = (__bf16*)alloc((size_t)ATTN_INNER * DIMX * 2);
  __bf16* wfo  = (__bf16*)alloc((size_t)FF_INNER * DIMX * 2);
  __bf16* xn   = (__bf16*)alloc((size_t)MROWS * DIMX * 2);
  float*  qf   = (float*) alloc((size_t)MROWS * ATTN_INNER * 4);
  float*  kf   = (float*) alloc((size_t)MROWS * DHEAD * 4);
  float*  vf   = (float*) alloc((size_t)MROWS * DHEAD * 4);
  __bf16* ffb  = (__bf16*)alloc((size_t)MROWS * 2 * FF_INNER * 2);
  __bf16* qr   = (__bf16*)alloc((size_t)MROWS * ATTN_INNER * 2);
  __bf16* kT   = (__bf16*)alloc((size_t)BATCH * DHEAD * NSEQ * 2);
  __bf16* vb   = (__bf16*)alloc((size_t)MROWS * DHEAD * 2);
  __bf16* attn = (__bf16*)alloc((size_t)MROWS * ATTN_INNER * 2);

  convert_bf16_kernel<<<2048, 256, 0, stream>>>(w_fused, wf, (size_t)DIMX * FUSEDN);
  convert_bf16_kernel<<<2048, 256, 0, stream>>>(w_attn_out, wao, (size_t)ATTN_INNER * DIMX);
  convert_bf16_kernel<<<2048, 256, 0, stream>>>(w_ff_out, wfo, (size_t)FF_INNER * DIMX);

  layernorm_kernel<<<MROWS, 256, 0, stream>>>(x, gamma, xn);

  gemm_fused_kernel<<<dim3(FUSEDN / 128, MROWS / 128), 256, 0, stream>>>(xn, wf, qf, kf, vf, ffb);

  rope_kernel<<<MROWS, 256, 0, stream>>>(qf, kf, vf, qr, kT, vb);

  attn_kernel<<<dim3(NSEQ / 64, HEADS, BATCH), 128, 0, stream>>>(qr, kT, vb, attn);

  swiglu_kernel<<<4096, 256, 0, stream>>>(ffb);

  gemm_bf16_kernel<<<dim3(DIMX / 128, MROWS / 128), 256, 0, stream>>>(
      attn, ATTN_INNER, wao, DIMX, out, DIMX, ATTN_INNER, 0);
  gemm_bf16_kernel<<<dim3(DIMX / 128, MROWS / 128), 256, 0, stream>>>(
      ffb, 2 * FF_INNER, wfo, DIMX, out, DIMX, FF_INNER, 1);
}
